// CpGenerateExport_71399536329352
// MI455X (gfx1250) — matrix operand order repaired and checked by reading, not compiled, not run on hardware
//
#include <hip/hip_runtime.h>

// ---------------- model constants ----------------
#define D_MODEL    1024
#define N_HEADS    16
#define HEAD_DIM   64
#define D_FF       4096
#define N_LAYERS   4
#define VOCAB      2048
#define NUM_GROUPS 15
#define CACHE_LEN  17

// ---------------- launch shape -------------------
#define NBLK   64
#define TPB    256
#define WPB    (TPB / 32)        // 8 waves per block (wave32)
#define GWAVES (NBLK * WPB)      // 512 waves total

typedef __bf16 bf16;
typedef __attribute__((ext_vector_type(16))) __bf16 v16bf;
typedef __attribute__((ext_vector_type(8)))  __bf16 v8bf;
typedef __attribute__((ext_vector_type(4)))  __bf16 v4bf;
typedef __attribute__((ext_vector_type(8)))  float  v8f;
typedef __attribute__((ext_vector_type(4)))  float  v4f;   // native vector for NT loads

// ---------------- device-wide barrier ----------------
// bar[0] = arrival count, bar[1] = generation; both zeroed by hipMemsetAsync
// before the kernel each call (graph-capture safe).
__device__ inline void grid_sync(unsigned* bar) {
  __threadfence();
  __syncthreads();
  if (threadIdx.x == 0) {
    unsigned gen = __hip_atomic_load(&bar[1], __ATOMIC_RELAXED, __HIP_MEMORY_SCOPE_AGENT);
    unsigned arrived = __hip_atomic_fetch_add(&bar[0], 1u, __ATOMIC_ACQ_REL, __HIP_MEMORY_SCOPE_AGENT);
    if (arrived == NBLK - 1) {
      __hip_atomic_store(&bar[0], 0u, __ATOMIC_RELAXED, __HIP_MEMORY_SCOPE_AGENT);
      __hip_atomic_fetch_add(&bar[1], 1u, __ATOMIC_RELEASE, __HIP_MEMORY_SCOPE_AGENT);
    } else {
      while (__hip_atomic_load(&bar[1], __ATOMIC_ACQUIRE, __HIP_MEMORY_SCOPE_AGENT) == gen) {
        __builtin_amdgcn_s_sleep(8);
      }
    }
  }
  __syncthreads();
  __threadfence();
}

// ---------------- fp32 -> bf16 weight conversion ----------------
// Non-temporal source loads: the fp32 image is read exactly once and must not
// evict the bf16 working set we are building in L2.
__device__ inline void cvt_bf16(const float* __restrict__ src, bf16* __restrict__ dst, size_t n) {
  const size_t start  = ((size_t)blockIdx.x * TPB + threadIdx.x) * 4;
  const size_t stride = (size_t)NBLK * TPB * 4;
  for (size_t i = start; i < n; i += stride) {
    v4f v = __builtin_nontemporal_load((const v4f*)(src + i));
    v4bf o;
    o[0] = (bf16)v[0]; o[1] = (bf16)v[1]; o[2] = (bf16)v[2]; o[3] = (bf16)v[3];
    *(v4bf*)(dst + i) = o;
  }
}

// ---------------- block reductions / RMSNorm ----------------
__device__ inline float block_sumsq(const float* x, float* s_red) {
  float ss = 0.f;
  for (int i = threadIdx.x; i < D_MODEL; i += TPB) { float v = x[i]; ss += v * v; }
  s_red[threadIdx.x] = ss;
  __syncthreads();
  for (int st = TPB / 2; st > 0; st >>= 1) {
    if (threadIdx.x < st) s_red[threadIdx.x] += s_red[threadIdx.x + st];
    __syncthreads();
  }
  float r = s_red[0];
  __syncthreads();
  return r;
}

__device__ inline void rms_to_bf16(const float* x, const float* w, bf16* dst, float* s_red) {
  float sc = rsqrtf(block_sumsq(x, s_red) * (1.0f / D_MODEL) + 1e-6f);
  for (int i = threadIdx.x; i < D_MODEL; i += TPB) dst[i] = (bf16)(x[i] * w[i] * sc);
  __syncthreads();
}

__device__ inline void rms_to_f32(const float* x, const float* w, float* dst, float* s_red) {
  float sc = rsqrtf(block_sumsq(x, s_red) * (1.0f / D_MODEL) + 1e-6f);
  for (int i = threadIdx.x; i < D_MODEL; i += TPB) dst[i] = x[i] * w[i] * sc;
  __syncthreads();
}

// ---------------- WMMA GEMV tile ----------------
// Computes y[n0..n0+15] = W[n0..n0+15][0..K) . x[0..K) with bf16 inputs and
// f32 accumulation using V_WMMA_F32_16X16X32_BF16.
//   A (16x32, bf16): activation vector lives in row M=0. Per ISA 7.12.2 the
//     16-bit A layout puts row 0 in lanes 0 and 16; here lane 0 holds K 0-15, lane 16 K 16-31.
//     All other lanes read a 32B zero pad in LDS (stride 0), so rows 1..15 stay
//     zero with no EXEC-mask manipulation in the loop.
//   B (32x16, bf16): column n = weight row n0+n. Lanes 0-15 hold K=0..15,
//     lanes 16-31 hold K=16..31 (contiguous runs, same k order as A).
//   D (16x16, f32): row M=0 outputs are acc[0] of lanes 0..15.
// The K loop is manually software-pipelined (ping-pong double buffer, compiler
// unrolling disabled): loads for chunk i+1 are in flight while WMMA consumes
// chunk i, keeping VGPR usage low (multi-wave occupancy) with partial waits.
// K must be a multiple of 64 (all call sites: 1024, 512, 128).
__device__ __forceinline__ float wmma_gemv16(const bf16* __restrict__ W, int ldw,
                                             const bf16* __restrict__ x, int K, int n0,
                                             const bf16* __restrict__ zpad) {
  const int lane = threadIdx.x & 31;
  const int hi   = lane >> 4;
  const bf16* wp = W + (size_t)(n0 + (lane & 15)) * ldw + hi * 16;
  const bool aLane = (lane & 15) == 0;       // lanes 0 and 16 carry A row 0
  const bf16* xp  = x + hi * 16;
  const bf16* ap0 = aLane ? xp      : zpad;  // zero-pad lanes walk with stride 0
  const bf16* ap1 = aLane ? xp + 8  : zpad;
  const int   ainc = aLane ? 32 : 0;
  v8f acc;
#pragma unroll
  for (int i = 0; i < 8; ++i) acc[i] = 0.0f;

  // preload chunk 0
  v16bf b0  = *(const v16bf*)(wp);
  v8bf  lo0 = *(const v8bf*)ap0;
  v8bf  hh0 = *(const v8bf*)ap1;

#pragma clang loop unroll(disable)
  for (int kb = 0; kb < K; kb += 64) {
    // issue loads for chunk kb+32 (always exists: K % 64 == 0)
    v16bf b1 = *(const v16bf*)(wp + kb + 32);
    ap0 += ainc; ap1 += ainc;
    v8bf lo1 = *(const v8bf*)ap0;
    v8bf hh1 = *(const v8bf*)ap1;

    v16bf a;
#pragma unroll
    for (int i = 0; i < 8; ++i) { a[i] = lo0[i]; a[i + 8] = hh0[i]; }
    acc = __builtin_amdgcn_wmma_f32_16x16x32_bf16(false, a, false, b0, (short)0, acc,
                                                  false, false);

    // issue loads for chunk kb+64 (next loop iteration), if any
    if (kb + 64 < K) {
      b0 = *(const v16bf*)(wp + kb + 64);
      ap0 += ainc; ap1 += ainc;
      lo0 = *(const v8bf*)ap0;
      hh0 = *(const v8bf*)ap1;
    }

#pragma unroll
    for (int i = 0; i < 8; ++i) { a[i] = lo1[i]; a[i + 8] = hh1[i]; }
    acc = __builtin_amdgcn_wmma_f32_16x16x32_bf16(false, a, false, b1, (short)0, acc,
                                                  false, false);
  }
  return acc[0];
}

// ---------------- per-block attention (cache is tiny: <=17 keys) ----------------
__device__ inline void attention_block(const float* __restrict__ q,
                                       const float* __restrict__ kcl,
                                       const float* __restrict__ vcl,
                                       int pos, float* s_p, bf16* s_abf) {
  const int wave = threadIdx.x >> 5;
  const int lane = threadIdx.x & 31;
  for (int h = wave; h < N_HEADS; h += WPB) {
    const float* qh = q + h * HEAD_DIM;
    const float* kh = kcl + h * CACHE_LEN * HEAD_DIM;
    const float* vh = vcl + h * CACHE_LEN * HEAD_DIM;
    float sc = -1e30f;
    if (lane <= pos) {                         // lane == cache position t
      const float* kr = kh + lane * HEAD_DIM;
      float d = 0.f;
#pragma unroll
      for (int i = 0; i < HEAD_DIM; i += 4) {
        v4f a = *(const v4f*)(qh + i);
        v4f b = *(const v4f*)(kr + i);
        d += a[0] * b[0] + a[1] * b[1] + a[2] * b[2] + a[3] * b[3];
      }
      sc = d * 0.125f;                         // 1/sqrt(64)
    }
    float mx = sc;
#pragma unroll
    for (int o = 16; o > 0; o >>= 1) mx = fmaxf(mx, __shfl_xor(mx, o));
    float e = (lane <= pos) ? __expf(sc - mx) : 0.f;
    float sm = e;
#pragma unroll
    for (int o = 16; o > 0; o >>= 1) sm += __shfl_xor(sm, o);
    if (lane <= pos) s_p[h * 20 + lane] = e / sm;
    // same-wave LDS accesses are processed in order, so reading p back is safe
    for (int d0 = lane; d0 < HEAD_DIM; d0 += 32) {
      float acc = 0.f;
      for (int t = 0; t <= pos; ++t) acc += s_p[h * 20 + t] * vh[t * HEAD_DIM + d0];
      s_abf[h * HEAD_DIM + d0] = (bf16)acc;
    }
  }
}

// ---------------- argmax (first-max, JAX semantics) + embedding gather ----------------
__device__ inline void argmax_gather(const float* __restrict__ logp,
                                     const float* __restrict__ etab,
                                     float* __restrict__ x, float* __restrict__ esum,
                                     float* s_red, int* s_redi) {
  float bv = -3.4e38f; int bi = 0;
  for (int i = threadIdx.x; i < VOCAB; i += TPB) {
    float v = logp[i];
    if (v > bv) { bv = v; bi = i; }            // ascending scan -> keeps lowest idx on tie
  }
  s_red[threadIdx.x] = bv; s_redi[threadIdx.x] = bi;
  __syncthreads();
  for (int st = TPB / 2; st > 0; st >>= 1) {
    if (threadIdx.x < st) {
      float ov = s_red[threadIdx.x + st]; int oi = s_redi[threadIdx.x + st];
      float mv = s_red[threadIdx.x];      int mi = s_redi[threadIdx.x];
      if (ov > mv || (ov == mv && oi < mi)) { s_red[threadIdx.x] = ov; s_redi[threadIdx.x] = oi; }
    }
    __syncthreads();
  }
  int code = s_redi[0];
  __syncthreads();
  const float* emb = etab + (size_t)code * D_MODEL;
  for (int i = threadIdx.x; i < D_MODEL; i += TPB) {
    float e = emb[i];
    x[i] = e;                                  // next step input (row 0)
    esum[i] += e;                              // embed_sum accumulator
  }
}

// ---------------- context ----------------
struct Ctx {
  const float *ln1, *ln2, *lnf;
  const bf16 *bq, *bk, *bv, *bo, *b1, *b2, *b3;
  float *kc, *vc, *x, *q, *esum, *af, *bfv, *wop, *w2p;
  unsigned* bar;
  bf16 *s_hbf, *s_abf, *s_g, *s_zero;
  float *s_hid, *s_p, *s_red;
  int   *s_redi;
};

// ---------------- one transformer step (S rows at positions pos0..pos0+S-1) ----------------
__device__ void tstep(const Ctx& c, int S, int pos0) {
  const int lane = threadIdx.x & 31;
  const int gw   = blockIdx.x * WPB + (threadIdx.x >> 5);
  for (int l = 0; l < N_LAYERS; ++l) {
    const bf16* Wq = c.bq + (size_t)l * D_MODEL * D_MODEL;
    const bf16* Wk = c.bk + (size_t)l * D_MODEL * D_MODEL;
    const bf16* Wv = c.bv + (size_t)l * D_MODEL * D_MODEL;
    const bf16* Wo = c.bo + (size_t)l * D_MODEL * D_MODEL;
    const bf16* W1 = c.b1 + (size_t)l * D_FF * D_MODEL;
    const bf16* W2 = c.b2 + (size_t)l * D_MODEL * D_FF;
    const bf16* W3 = c.b3 + (size_t)l * D_FF * D_MODEL;
    float* kcl = c.kc + l * N_HEADS * CACHE_LEN * HEAD_DIM;
    float* vcl = c.vc + l * N_HEADS * CACHE_LEN * HEAD_DIM;

    // ---- Phase 1: RMSNorm(ln1) + fused QKV GEMV + RoPE + KV-cache write ----
    for (int s = 0; s < S; ++s) {
      const int pos = pos0 + s;
      rms_to_bf16(c.x + s * D_MODEL, c.ln1 + l * D_MODEL, c.s_hbf, c.s_red);
      for (int t = gw; t < 192; t += GWAVES) {   // 3 * (1024/16) tiles
        const int m  = t >> 6;                   // 0:q 1:k 2:v
        const int n0 = (t & 63) << 4;
        const bf16* W = (m == 0) ? Wq : (m == 1) ? Wk : Wv;
        float val = wmma_gemv16(W, D_MODEL, c.s_hbf, D_MODEL, n0, c.s_zero);
        float res = val;
        if (m < 2) {                             // RoPE on q,k
          float other = __shfl_xor(val, 1);
          int d = n0 + lane;
          float j2  = (float)((d & 63) & ~1);
          float inv = __expf(-9.210340371976184f * j2 * (1.0f / 64.0f)); // 10000^(-j2/64)
          float ang = (float)pos * inv;
          float cs = __cosf(ang), sn = __sinf(ang);
          res = (d & 1) ? (other * sn + val * cs) : (val * cs - other * sn);
        }
        if (lane < 16) {
          int d = n0 + lane;
          if (m == 0)      c.q[s * D_MODEL + d] = res;
          else if (m == 1) kcl[((d >> 6) * CACHE_LEN + pos) * HEAD_DIM + (d & 63)] = res;
          else             vcl[((d >> 6) * CACHE_LEN + pos) * HEAD_DIM + (d & 63)] = val;
        }
      }
      __syncthreads();
    }
    grid_sync(c.bar);

    // ---- Phase 2: attention (block-replicated, cache is tiny) + WO partials (K split 8x128) ----
    for (int s = 0; s < S; ++s) {
      const int pos = pos0 + s;
      attention_block(c.q + s * D_MODEL, kcl, vcl, pos, c.s_p, c.s_abf);
      __syncthreads();
      for (int t = gw; t < 512; t += GWAVES) {
        const int n0 = (t & 63) << 4;
        const int kc = t >> 6;
        float val = wmma_gemv16(Wo + kc * 128, D_MODEL, c.s_abf + kc * 128, 128, n0, c.s_zero);
        if (lane < 16) c.wop[(s * 8 + kc) * D_MODEL + n0 + lane] = val;
      }
      __syncthreads();
    }
    grid_sync(c.bar);

    // ---- Phase 3: merged-x RMSNorm(ln2) + W1/W3 GEMV ----
    for (int s = 0; s < S; ++s) {
      for (int i = threadIdx.x; i < D_MODEL; i += TPB) {
        float acc = c.x[s * D_MODEL + i];
#pragma unroll
        for (int cc = 0; cc < 8; ++cc) acc += c.wop[(s * 8 + cc) * D_MODEL + i];
        c.s_hid[i] = acc;                        // x + attn_out (kept virtual until Phase 5)
      }
      __syncthreads();
      rms_to_bf16(c.s_hid, c.ln2 + l * D_MODEL, c.s_hbf, c.s_red);
      for (int t = gw; t < 512; t += GWAVES) {   // 256 tiles W1 + 256 tiles W3
        const int which = t >> 8;
        const int n0    = (t & 255) << 4;
        const bf16* W = which ? W3 : W1;
        float val = wmma_gemv16(W, D_MODEL, c.s_hbf, D_MODEL, n0, c.s_zero);
        if (lane < 16) (which ? c.bfv : c.af)[s * D_FF + n0 + lane] = val;
      }
      __syncthreads();
    }
    grid_sync(c.bar);

    // ---- Phase 4: silu(a)*b into LDS + W2 partials (K split 8x512) ----
    for (int s = 0; s < S; ++s) {
      for (int i = threadIdx.x; i < D_FF; i += TPB) {
        float a = c.af[s * D_FF + i];
        float b = c.bfv[s * D_FF + i];
        c.s_g[i] = (bf16)((a / (1.0f + __expf(-a))) * b);
      }
      __syncthreads();
      for (int t = gw; t < 512; t += GWAVES) {
        const int n0 = (t & 63) << 4;
        const int kc = t >> 6;
        float val = wmma_gemv16(W2 + kc * 512, D_FF, c.s_g + kc * 512, 512, n0, c.s_zero);
        if (lane < 16) c.w2p[(s * 8 + kc) * D_MODEL + n0 + lane] = val;
      }
      __syncthreads();
    }
    grid_sync(c.bar);

    // ---- Phase 5: materialize x += attn_out + ffn_out ----
    {
      const int gtid = blockIdx.x * TPB + threadIdx.x;
      for (int i = gtid; i < S * D_MODEL; i += NBLK * TPB) {
        const int s = i >> 10, d = i & (D_MODEL - 1);
        float acc = c.x[i];
#pragma unroll
        for (int cc = 0; cc < 8; ++cc)
          acc += c.wop[(s * 8 + cc) * D_MODEL + d] + c.w2p[(s * 8 + cc) * D_MODEL + d];
        c.x[i] = acc;
      }
    }
    grid_sync(c.bar);
  }
}

// ---------------- megakernel ----------------
extern "C" __global__ void __launch_bounds__(TPB)
talker_mega(const float* __restrict__ th,  const float* __restrict__ c0e,
            const float* __restrict__ wq,  const float* __restrict__ wk,
            const float* __restrict__ wv,  const float* __restrict__ wo,
            const float* __restrict__ w1,  const float* __restrict__ w2,
            const float* __restrict__ w3,
            const float* __restrict__ ln1, const float* __restrict__ ln2,
            const float* __restrict__ lnf,
            const float* __restrict__ heads, const float* __restrict__ embeds,
            float* __restrict__ out, unsigned char* __restrict__ ws)
{
  __shared__ __align__(16) bf16  s_hbf[D_MODEL];
  __shared__ __align__(16) bf16  s_abf[D_MODEL];
  __shared__ __align__(16) bf16  s_g[D_FF];
  __shared__ __align__(16) bf16  s_zero[16];   // 32B zero pad for WMMA A lanes
  __shared__ __align__(16) float s_hid[D_MODEL];
  __shared__ float s_p[N_HEADS * 20];
  __shared__ float s_red[TPB];
  __shared__ int   s_redi[TPB];

  if (threadIdx.x < 16) s_zero[threadIdx.x] = (bf16)0.0f;
  __syncthreads();

  const size_t MQ = (size_t)N_LAYERS * D_MODEL * D_MODEL;   // 4M elems
  const size_t MF = (size_t)N_LAYERS * D_FF * D_MODEL;      // 16M elems
  unsigned* bar = (unsigned*)ws;                            // zeroed by memset
  bf16* bq = (bf16*)(ws + 256);
  bf16* bk = bq + MQ;  bf16* bv = bk + MQ;  bf16* bo = bv + MQ;
  bf16* b1 = bo + MQ;  bf16* b2 = b1 + MF;  bf16* b3 = b2 + MF;
  const int CSZ = N_LAYERS * N_HEADS * CACHE_LEN * HEAD_DIM;
  float* kcache = (float*)(b3 + MF);
  float* vcache = kcache + CSZ;
  float* xbuf   = vcache + CSZ;          // [2][1024] residual stream
  float* qbuf   = xbuf + 2 * D_MODEL;    // [2][1024]
  float* esum   = qbuf + 2 * D_MODEL;    // [1024]
  float* af     = esum + D_MODEL;        // [2][4096] w1 out
  float* bfv    = af + 2 * D_FF;         // [2][4096] w3 out
  float* wop    = bfv + 2 * D_FF;        // [2][8][1024] wo partials
  float* w2p    = wop + 16 * D_MODEL;    // [2][8][1024] w2 partials

  // Phase 0: fp32 -> bf16 weight conversion (makes 128MB working set L2-resident) + init
  cvt_bf16(wq, bq, MQ); cvt_bf16(wk, bk, MQ);
  cvt_bf16(wv, bv, MQ); cvt_bf16(wo, bo, MQ);
  cvt_bf16(w1, b1, MF); cvt_bf16(w2, b2, MF); cvt_bf16(w3, b3, MF);
  if (blockIdx.x == 0) {
    for (int i = threadIdx.x; i < D_MODEL; i += TPB) {
      xbuf[i]           = th[i];           // row 0 = talker_hidden
      xbuf[D_MODEL + i] = c0e[i];          // row 1 = code_0_embed
      esum[i]           = c0e[i];          // embed_sum starts at code_0_embed
    }
  }
  grid_sync(bar);

  Ctx c;
  c.ln1 = ln1; c.ln2 = ln2; c.lnf = lnf;
  c.bq = bq; c.bk = bk; c.bv = bv; c.bo = bo; c.b1 = b1; c.b2 = b2; c.b3 = b3;
  c.kc = kcache; c.vc = vcache; c.x = xbuf; c.q = qbuf; c.esum = esum;
  c.af = af; c.bfv = bfv; c.wop = wop; c.w2p = w2p; c.bar = bar;
  c.s_hbf = s_hbf; c.s_abf = s_abf; c.s_g = s_g; c.s_zero = s_zero;
  c.s_hid = s_hid; c.s_p = s_p; c.s_red = s_red; c.s_redi = s_redi;

  // initial step: S=2 at positions {0,1}
  tstep(c, 2, 0);

  const int lane = threadIdx.x & 31;
  const int gw   = blockIdx.x * WPB + (threadIdx.x >> 5);

  for (int g = 0; g < NUM_GROUPS; ++g) {
    const int row = (g == 0) ? 1 : 0;          // hidden = lnf-normed last row
    // logits = rmsnorm(x,lnf) @ heads[g].T  (heads fp32, streamed non-temporal:
    // read-once data must not evict the bf16 weights from L2)
    rms_to_f32(xbuf + row * D_MODEL, lnf, s_hid, s_red);
    const float* hg = heads + (size_t)g * VOCAB * D_MODEL;
    for (int o = gw; o < VOCAB; o += GWAVES) {
      const float* hr = hg + (size_t)o * D_MODEL;
      float acc = 0.f;
      for (int i = lane * 4; i < D_MODEL; i += 128) {
        v4f a = __builtin_nontemporal_load((const v4f*)(hr + i));
        acc += a[0] * s_hid[i] + a[1] * s_hid[i + 1] + a[2] * s_hid[i + 2] + a[3] * s_hid[i + 3];
      }
#pragma unroll
      for (int off = 16; off > 0; off >>= 1) acc += __shfl_xor(acc, off);
      if (lane == 0) out[g * VOCAB + o] = acc;
    }
    grid_sync(bar);
    // argmax -> code -> embedding gather -> next-step input + embed_sum
    if (blockIdx.x == 0)
      argmax_gather(out + g * VOCAB, embeds + (size_t)g * VOCAB * D_MODEL,
                    xbuf, esum, s_red, s_redi);
    grid_sync(bar);
    // reference discards the final step's hidden state -> skip the 16th step
    if (g < NUM_GROUPS - 1) tstep(c, 1, 2 + g);
  }

  if (blockIdx.x == 0)
    for (int i = threadIdx.x; i < D_MODEL; i += TPB)
      out[NUM_GROUPS * VOCAB + i] = esum[i];
}

// ---------------- host entry ----------------
extern "C" void kernel_launch(void* const* d_in, const int* in_sizes, int n_in,
                              void* d_out, int out_size, void* d_ws, size_t ws_size,
                              hipStream_t stream) {
  (void)in_sizes; (void)n_in; (void)out_size; (void)ws_size;
  // zero the grid-barrier words (stream memset: graph-capture legal)
  (void)hipMemsetAsync(d_ws, 0, 256, stream);
  talker_mega<<<dim3(NBLK), dim3(TPB), 0, stream>>>(
      (const float*)d_in[0],  (const float*)d_in[1],
      (const float*)d_in[2],  (const float*)d_in[3],  (const float*)d_in[4],
      (const float*)d_in[5],  (const float*)d_in[6],  (const float*)d_in[7],
      (const float*)d_in[8],  (const float*)d_in[9],  (const float*)d_in[10],
      (const float*)d_in[11], (const float*)d_in[12], (const float*)d_in[13],
      (float*)d_out, (unsigned char*)d_ws);
}
